// MultiHeadSelectiveScan_39865886441840
// MI455X (gfx1250) — compile-verified, matrix operand order checked by reading, not run on hardware
//
#include <hip/hip_runtime.h>
#include <hip/hip_bf16.h>
#include <stdint.h>

// ---------------- problem constants ----------------
#define DIMC   256      // model dim C == d_inner
#define NSTATE 16
#define DTRANK 16
#define NPROJ  48       // dt_rank + 2*d_state
#define BATCH  4
#define LSEQ   2304     // 48*48
#define BLM    (BATCH*LSEQ)   // 9216 rows (multiple of 128)
#define NIN    512      // 2*d_inner
#define CL     64       // scan chunk length
#define NCH    (LSEQ/CL)      // 36 chunks per sequence

typedef _Float16 half16 __attribute__((ext_vector_type(16)));
typedef float    float8 __attribute__((ext_vector_type(8)));

// ---------------- workspace layout (bytes, all 256B aligned) ----------------
static const size_t OFF_XH   = 0;                               // BLM*256 f16
static const size_t OFF_WIN  = OFF_XH   + (size_t)BLM*DIMC*2;   // 512*256 f16
static const size_t OFF_WOUT = OFF_WIN  + (size_t)NIN*DIMC*2;   // 256*256 f16
static const size_t OFF_XZ   = OFF_WOUT + (size_t)DIMC*DIMC*2;  // BLM*512 f32
static const size_t OFF_XS   = OFF_XZ   + (size_t)BLM*NIN*4;    // 2*BLM*256 f32
static const size_t OFF_XDBL = OFF_XS   + (size_t)2*BLM*DIMC*4; // 2*BLM*48 f32
static const size_t OFF_Y    = OFF_XDBL + (size_t)2*BLM*NPROJ*4;// 2*BLM*256 f32
static const size_t OFF_YCH  = OFF_Y    + (size_t)2*BLM*DIMC*4; // BLM*256 f16
static const size_t OFF_DT   = OFF_YCH  + (size_t)BLM*DIMC*2;   // 2*BLM*256 f32
static const size_t SZ_PQ    = (size_t)2*BATCH*NCH*DIMC*NSTATE*4; // 4.7MB
static const size_t OFF_P    = OFF_DT   + (size_t)2*BLM*DIMC*4;
static const size_t OFF_Q    = OFF_P    + SZ_PQ;
static const size_t OFF_HST  = OFF_Q    + SZ_PQ;

// ---------------- WMMA fragment helpers ----------------
union FragAB {
    uint4  q[2];
    half16 h;
};

// Row-major f16 matrix fragment. Wave32 16-bit A/B layout: lane%16 = M (or N),
// lane/16 = K-half; VGPR0..3 K = kh*8+{0..7}; VGPR4..7 K = 16+kh*8+{0..7}.
__device__ __forceinline__ void load_frag(const _Float16* rowBase, int k0, int kh, FragAB& f) {
    const char* p = (const char*)(rowBase + k0 + kh * 8);
    f.q[0] = *(const uint4*)(p);
    f.q[1] = *(const uint4*)(p + 32);
}

// One wave computes a 16(M) x 64(N) f32 tile: A frag reused across 4 B frags.
__device__ __forceinline__ void gemm_wave_16x64(const _Float16* A, const _Float16* Bt,
                                                int lda, int ldb, int K,
                                                int m0, int n0, float8 acc[4]) {
    const int lane = threadIdx.x & 31;
    const int ml   = lane & 15;
    const int kh   = lane >> 4;
    const _Float16* arow  = A  + (size_t)(m0 + ml) * lda;
    const _Float16* brow0 = Bt + (size_t)(n0 + ml) * ldb;
    for (int k0 = 0; k0 < K; k0 += 32) {
        FragAB a;
        load_frag(arow, k0, kh, a);
#pragma unroll
        for (int s = 0; s < 4; ++s) {
            FragAB b;
            load_frag(brow0 + (size_t)(s * 16) * ldb, k0, kh, b);
            acc[s] = __builtin_amdgcn_wmma_f32_16x16x32_f16(
                false, a.h, false, b.h, (short)0, acc[s], false, false);
        }
    }
}

// ---------------- prep kernels ----------------
__global__ void k_convert_weights(const float* w_in, const float* w_out, _Float16* ws_base) {
    const size_t i = (size_t)blockIdx.x * blockDim.x + threadIdx.x;
    _Float16* win  = ws_base;
    _Float16* wout = ws_base + (size_t)NIN * DIMC;
    const size_t nin = (size_t)NIN * DIMC;
    const size_t tot = nin + (size_t)DIMC * DIMC;
    if (i < nin)       win[i]        = (_Float16)w_in[i];
    else if (i < tot)  wout[i - nin] = (_Float16)w_out[i - nin];
}

// x (B,C,H,W) f32 -> xh (B, L, C) f16
__global__ void k_transpose_x(const float* x, _Float16* xh) {
    const size_t i = (size_t)blockIdx.x * blockDim.x + threadIdx.x;
    if (i >= (size_t)BLM * DIMC) return;
    const int c = (int)(i & (DIMC - 1));
    const size_t bl = i >> 8;
    const int l = (int)(bl % LSEQ);
    const int b = (int)(bl / LSEQ);
    xh[i] = (_Float16)x[((size_t)b * DIMC + c) * LSEQ + l];
}

// ---------------- GEMM kernels ----------------
__global__ void k_gemm_in(const _Float16* xh, const _Float16* w, float* xz) {
    const int wid = threadIdx.x >> 5;
    const int m0  = blockIdx.x * 128 + wid * 16;
    const int n0  = blockIdx.y * 64;
    float8 acc[4];
#pragma unroll
    for (int s = 0; s < 4; ++s) acc[s] = (float8)0.0f;
    gemm_wave_16x64(xh, w, DIMC, DIMC, DIMC, m0, n0, acc);
    const int lane = threadIdx.x & 31;
    const int ml = lane & 15, kh = lane >> 4;
#pragma unroll
    for (int s = 0; s < 4; ++s)
#pragma unroll
        for (int r = 0; r < 8; ++r) {
            const int m = m0 + r + kh * 8;
            const int n = n0 + s * 16 + ml;
            xz[(size_t)m * NIN + n] = acc[s][r];
        }
}

__global__ void k_gemm_out(const _Float16* ych, const _Float16* w,
                           const float* scale_mod, float* out) {
    const int wid = threadIdx.x >> 5;
    const int m0  = blockIdx.x * 128 + wid * 16;
    const int n0  = blockIdx.y * 64;
    float8 acc[4];
#pragma unroll
    for (int s = 0; s < 4; ++s) acc[s] = (float8)0.0f;
    gemm_wave_16x64(ych, w, DIMC, DIMC, DIMC, m0, n0, acc);
    const float sc = scale_mod[0];
    const int lane = threadIdx.x & 31;
    const int ml = lane & 15, kh = lane >> 4;
#pragma unroll
    for (int s = 0; s < 4; ++s)
#pragma unroll
        for (int r = 0; r < 8; ++r) {
            const int m = m0 + r + kh * 8;
            const int n = n0 + s * 16 + ml;
            const int b = m / LSEQ, l = m % LSEQ;
            out[((size_t)b * DIMC + n) * LSEQ + l] = acc[s][r] * sc;
        }
}

// ---------------- causal depthwise conv + SiLU (both directions) ----------------
__global__ void k_conv_silu(const float* xz, const float* conv_w, const float* conv_b,
                            float* xs) {
    const size_t idx = (size_t)blockIdx.x * blockDim.x + threadIdx.x;
    if (idx >= (size_t)2 * BLM * DIMC) return;
    const int d = (int)(idx & (DIMC - 1));
    size_t t = idx >> 8;
    const int l = (int)(t % LSEQ); t /= LSEQ;
    const int b = (int)(t % BATCH);
    const int dir = (int)(t / BATCH);
    float acc = conv_b[d];
    const size_t rowb = (size_t)b * LSEQ;
    if (dir == 0) {
#pragma unroll
        for (int j = 0; j < 4; ++j) {
            const int lw = l - 3 + j;
            if (lw >= 0) acc += conv_w[d * 4 + j] * xz[(rowb + lw) * NIN + d];
        }
    } else {
#pragma unroll
        for (int j = 0; j < 4; ++j) {
            const int lw = l + 3 - j;
            if (lw < LSEQ) acc += conv_w[d * 4 + j] * xz[(rowb + lw) * NIN + d];
        }
    }
    xs[idx] = acc / (1.0f + __expf(-acc));   // SiLU
}

// ---------------- x_proj ----------------
__global__ void k_xproj(const float* xs, const float* xpw, float* xdbl) {
    const size_t idx = (size_t)blockIdx.x * blockDim.x + threadIdx.x;
    if (idx >= (size_t)2 * BLM * NPROJ) return;
    const int e = (int)(idx % NPROJ);
    const size_t row = idx / NPROJ;
    const float4* xr = (const float4*)(xs + row * DIMC);
    const float4* wr = (const float4*)(xpw + (size_t)e * DIMC);
    float acc = 0.f;
#pragma unroll 4
    for (int c = 0; c < DIMC / 4; ++c) {
        const float4 a = xr[c], b = wr[c];
        acc += a.x * b.x + a.y * b.y + a.z * b.z + a.w * b.w;
    }
    xdbl[idx] = acc;
}

// ---------------- dt = softplus(dt_r . W^T + b), one block per row ----------------
__global__ void k_dt(const float* xdbl, const float* dtw_g, const float* dtb_g,
                     float* dtbuf) {
    const size_t row = blockIdx.x;        // dir*BLM + b*L + l  (2*BLM rows)
    const int d = threadIdx.x;
    __shared__ float sr[DTRANK];
    if (d < DTRANK) sr[d] = xdbl[row * NPROJ + d];
    __syncthreads();
    float dt = dtb_g[d];
#pragma unroll
    for (int r = 0; r < DTRANK; ++r) dt += sr[r] * dtw_g[d * DTRANK + r];
    dt = (dt > 20.0f) ? dt : log1pf(__expf(dt));
    dtbuf[row * DIMC + d] = dt;
}

// ---------------- chunk-parallel selective scan ----------------
// Phase A: per-chunk summaries  P = prod(dA),  Q = h_end starting from h=0.
__global__ void k_scan_chunk_a(const float* dtbuf, const float* xs, const float* xdbl,
                               const float* Alog, float* Pbuf, float* Qbuf) {
    const int blk = blockIdx.x;           // ((dir*4+b)*NCH + c)
    const int c   = blk % NCH;
    const int g   = blk / NCH;            // dir*4+b
    const int b   = g & 3, dir = g >> 2;
    const int d   = threadIdx.x;
    const size_t rowbase = (size_t)dir * BLM + (size_t)b * LSEQ;

    __shared__ float sB[CL * NSTATE];     // B rows for this chunk
    for (int idx = d; idx < CL * NSTATE; idx += 256) {
        const int i = idx >> 4, j = idx & 15;
        const int t = c * CL + i;
        const int p = dir ? (LSEQ - 1 - t) : t;
        sB[idx] = xdbl[(rowbase + p) * NPROJ + DTRANK + j];
    }
    __syncthreads();

    float Ar[NSTATE], h[NSTATE], P[NSTATE];
#pragma unroll
    for (int n = 0; n < NSTATE; ++n) {
        Ar[n] = -__expf(Alog[d * NSTATE + n]);
        h[n] = 0.f; P[n] = 1.f;
    }
    for (int i = 0; i < CL; ++i) {
        const int t = c * CL + i;
        const int p = dir ? (LSEQ - 1 - t) : t;
        const float dt = dtbuf[(rowbase + p) * DIMC + d];
        const float xv = xs[(rowbase + p) * DIMC + d];
#pragma unroll
        for (int n = 0; n < NSTATE; ++n) {
            const float dA = __expf(dt * Ar[n]);
            h[n] = dA * h[n] + (dt * sB[i * NSTATE + n]) * xv;
            P[n] *= dA;
        }
    }
    const size_t base = ((size_t)blk * DIMC + d) * NSTATE;
#pragma unroll
    for (int n = 0; n < NSTATE; ++n) { Pbuf[base + n] = P[n]; Qbuf[base + n] = h[n]; }
}

// Phase B: sequential combine of chunk summaries -> h_start per chunk (36 steps).
__global__ void k_scan_prop(const float* Pbuf, const float* Qbuf, float* Hst) {
    const int g = blockIdx.x;             // dir*4+b, 8 blocks
    const int d = threadIdx.x;
    float h[NSTATE];
#pragma unroll
    for (int n = 0; n < NSTATE; ++n) h[n] = 0.f;
    for (int c = 0; c < NCH; ++c) {
        const size_t base = (((size_t)g * NCH + c) * DIMC + d) * NSTATE;
#pragma unroll
        for (int n = 0; n < NSTATE; ++n) {
            Hst[base + n] = h[n];
            h[n] = Pbuf[base + n] * h[n] + Qbuf[base + n];
        }
    }
}

// Phase C: replay each chunk from its true h_start, emit y with epilogue.
__global__ void k_scan_chunk_c(const float* dtbuf, const float* xs, const float* xdbl,
                               const float* xz, const float* Alog, const float* Dp,
                               const float* Hst, float* ybuf) {
    const int blk = blockIdx.x;
    const int c   = blk % NCH;
    const int g   = blk / NCH;
    const int b   = g & 3, dir = g >> 2;
    const int d   = threadIdx.x;
    const size_t rowbase = (size_t)dir * BLM + (size_t)b * LSEQ;

    __shared__ float sBC[CL * 32];        // B(16) | C(16) per step
    for (int idx = d; idx < CL * 32; idx += 256) {
        const int i = idx >> 5, j = idx & 31;
        const int t = c * CL + i;
        const int p = dir ? (LSEQ - 1 - t) : t;
        sBC[idx] = xdbl[(rowbase + p) * NPROJ + DTRANK + j];
    }
    __syncthreads();

    float Ar[NSTATE], h[NSTATE];
    const size_t hbase = ((size_t)blk * DIMC + d) * NSTATE;
#pragma unroll
    for (int n = 0; n < NSTATE; ++n) {
        Ar[n] = -__expf(Alog[d * NSTATE + n]);
        h[n] = Hst[hbase + n];
    }
    const float Dd = Dp[d];
    for (int i = 0; i < CL; ++i) {
        const int t = c * CL + i;
        const int p = dir ? (LSEQ - 1 - t) : t;
        const float dt = dtbuf[(rowbase + p) * DIMC + d];
        const float xv = xs[(rowbase + p) * DIMC + d];
        float y = 0.f;
#pragma unroll
        for (int n = 0; n < NSTATE; ++n) {
            const float dA = __expf(dt * Ar[n]);
            h[n] = dA * h[n] + (dt * sBC[i * 32 + n]) * xv;
            y += h[n] * sBC[i * 32 + 16 + n];
        }
        y += xv * Dd;
        const float zv = xz[((size_t)b * LSEQ + p) * NIN + DIMC + d];
        y *= zv / (1.0f + __expf(-zv));   // * silu(z)
        ybuf[(rowbase + p) * DIMC + d] = y;
    }
}

// ---------------- combine fwd+bwd -> f16 ----------------
__global__ void k_combine(const float* y, _Float16* ych) {
    const size_t i = (size_t)blockIdx.x * blockDim.x + threadIdx.x;
    if (i >= (size_t)BLM * DIMC) return;
    ych[i] = (_Float16)(y[i] + y[i + (size_t)BLM * DIMC]);
}

// ---------------- host launch ----------------
extern "C" void kernel_launch(void* const* d_in, const int* in_sizes, int n_in,
                              void* d_out, int out_size, void* d_ws, size_t ws_size,
                              hipStream_t stream) {
    const float* x        = (const float*)d_in[0];
    const float* in_w     = (const float*)d_in[1];
    const float* conv_w   = (const float*)d_in[2];
    const float* conv_b   = (const float*)d_in[3];
    const float* xproj_w  = (const float*)d_in[4];
    const float* dtproj_w = (const float*)d_in[5];
    const float* dtproj_b = (const float*)d_in[6];
    const float* A_log    = (const float*)d_in[7];
    const float* D_param  = (const float*)d_in[8];
    const float* out_w    = (const float*)d_in[9];
    const float* scale    = (const float*)d_in[10];
    float* out = (float*)d_out;

    char* ws = (char*)d_ws;
    _Float16* xh    = (_Float16*)(ws + OFF_XH);
    _Float16* win_h = (_Float16*)(ws + OFF_WIN);
    _Float16* wout_h= (_Float16*)(ws + OFF_WOUT);
    float*    xz    = (float*)   (ws + OFF_XZ);
    float*    xs    = (float*)   (ws + OFF_XS);
    float*    xdbl  = (float*)   (ws + OFF_XDBL);
    float*    ybuf  = (float*)   (ws + OFF_Y);
    _Float16* ych   = (_Float16*)(ws + OFF_YCH);
    float*    dtbuf = (float*)   (ws + OFF_DT);
    float*    Pbuf  = (float*)   (ws + OFF_P);
    float*    Qbuf  = (float*)   (ws + OFF_Q);
    float*    Hst   = (float*)   (ws + OFF_HST);

    const int TB = 256;
    // 0) conversions
    k_convert_weights<<<(196608 + TB - 1) / TB, TB, 0, stream>>>(in_w, out_w, win_h);
    k_transpose_x<<<((size_t)BLM * DIMC + TB - 1) / TB, TB, 0, stream>>>(x, xh);
    // 1) in_proj GEMM (shared by both directions)
    k_gemm_in<<<dim3(BLM / 128, NIN / 64), TB, 0, stream>>>(xh, win_h, xz);
    // 2) causal depthwise conv + SiLU
    k_conv_silu<<<((size_t)2 * BLM * DIMC + TB - 1) / TB, TB, 0, stream>>>(xz, conv_w, conv_b, xs);
    // 3) x_proj
    k_xproj<<<((size_t)2 * BLM * NPROJ + TB - 1) / TB, TB, 0, stream>>>(xs, xproj_w, xdbl);
    // 4) dt projection + softplus (one block per (dir,b,l) row)
    k_dt<<<2 * BLM, TB, 0, stream>>>(xdbl, dtproj_w, dtproj_b, dtbuf);
    // 5) chunk-parallel selective scan
    k_scan_chunk_a<<<2 * BATCH * NCH, TB, 0, stream>>>(dtbuf, xs, xdbl, A_log, Pbuf, Qbuf);
    k_scan_prop<<<2 * BATCH, TB, 0, stream>>>(Pbuf, Qbuf, Hst);
    k_scan_chunk_c<<<2 * BATCH * NCH, TB, 0, stream>>>(dtbuf, xs, xdbl, xz, A_log, D_param, Hst, ybuf);
    // 6) combine directions -> f16
    k_combine<<<((size_t)BLM * DIMC + TB - 1) / TB, TB, 0, stream>>>(ybuf, ych);
    // 7) out_proj GEMM + scale + transpose to (B,C,H,W)
    k_gemm_out<<<dim3(BLM / 128, DIMC / 64), TB, 0, stream>>>(ych, wout_h, scale, out);
    (void)in_sizes; (void)n_in; (void)out_size; (void)ws_size;
}
